// CompactSPAttn_31731218383351
// MI455X (gfx1250) — matrix operand order repaired and checked by reading, not compiled, not run on hardware
//
#include <hip/hip_runtime.h>
#include <hip/hip_bf16.h>

// Flash attention forward, B=1 S=4096 H=16 D=128, fp32 in/out.
// (a) one-time K/V fp32->bf16 pre-pass into d_ws (removes per-block cvt VALU,
//     halves bytes; 32MB bf16 K/V is L2-resident on MI455X's 192MB L2),
// (b) double-buffered async global->LDS staging shared by all 8 waves (ASYNCcnt),
// (c) bf16 WMMA (v_wmma_f32_16x16x32_bf16) with the S^T = K.Q^T trick so the
//     score accumulator layout IS the P A-fragment layout (no cross-lane moves).

typedef __attribute__((ext_vector_type(16))) __bf16 bf16x16;
typedef __attribute__((ext_vector_type(8)))  float  f32x8;
typedef int v4i __attribute__((vector_size(16)));   // 128-bit async payload type

#define WMMA_BF16(A, B, C) \
  __builtin_amdgcn_wmma_f32_16x16x32_bf16(false, (A), false, (B), (short)0, (C), false, false)

constexpr int   kS     = 4096;
constexpr int   kH     = 16;
constexpr int   kD     = 128;
constexpr int   kWaves = 8;            // waves per workgroup (wave32)
constexpr int   kQTile = kWaves * 16;  // q rows per workgroup
constexpr float kScale = 0.08838834764831845f * 1.44269504088896340f; // D^-0.5 * log2e

union Frag { bf16x16 v; uint4 u[2]; };

// ---------------- async global -> LDS staging (CDNA5 ASYNCcnt path) ----------------
#if __has_builtin(__builtin_amdgcn_global_load_async_to_lds_b128)
#define FA_ASYNC 1
#else
#define FA_ASYNC 0
#endif

__device__ __forceinline__ void cp16(const __bf16* g, __bf16* l) {
#if FA_ASYNC
  __builtin_amdgcn_global_load_async_to_lds_b128(
      (__attribute__((address_space(1))) v4i*)g,
      (__attribute__((address_space(3))) v4i*)l, 0, 0);
#else
  *(uint4*)l = *(const uint4*)g;   // sync fallback through VGPRs
#endif
}

__device__ __forceinline__ void wait_async_le(int which /*0 or 4, compile-time*/) {
#if FA_ASYNC
#if __has_builtin(__builtin_amdgcn_s_wait_asynccnt)
  if (which == 0) __builtin_amdgcn_s_wait_asynccnt(0);
  else            __builtin_amdgcn_s_wait_asynccnt(4);
#else
  if (which == 0) asm volatile("s_wait_asynccnt 0x0" ::: "memory");
  else            asm volatile("s_wait_asynccnt 0x4" ::: "memory");
#endif
#endif
  (void)which;
}

// Stage one 32x128 bf16 K block + V block into LDS (8KB each). 256 threads,
// 16B chunks: 2 K-chunks + 2 V-chunks per thread -> 4 outstanding async ops.
__device__ __forceinline__ void stage_kv(const __bf16* __restrict__ Kb,
                                         const __bf16* __restrict__ Vb,
                                         __bf16* kdst, __bf16* vdst,
                                         int s0, int head, int t) {
#pragma unroll
  for (int it = 0; it < 2; ++it) {
    const int e   = (t + it * 256) * 8;       // element index in 32x128 block
    const int row = e >> 7, d = e & (kD - 1);
    const size_t go = ((size_t)(s0 + row) * kH + head) * kD + d;
    cp16(Kb + go, kdst + e);
    cp16(Vb + go, vdst + e);
  }
}

// ---------------- pre-pass: fp32 -> bf16 (8 elements / thread) ----------------
__global__ __launch_bounds__(256) void cvt_bf16_kernel(const float* __restrict__ in,
                                                       __bf16* __restrict__ out, int n8) {
  const int i = blockIdx.x * 256 + threadIdx.x;
  if (i < n8) {
    const float4* p = (const float4*)in + (size_t)i * 2;
    const float4 a = p[0], b = p[1];
    union { uint4 u; __bf16 h[8]; } r;
    r.h[0] = (__bf16)a.x; r.h[1] = (__bf16)a.y; r.h[2] = (__bf16)a.z; r.h[3] = (__bf16)a.w;
    r.h[4] = (__bf16)b.x; r.h[5] = (__bf16)b.y; r.h[6] = (__bf16)b.z; r.h[7] = (__bf16)b.w;
    ((uint4*)out)[i] = r.u;
  }
}

// ---------------- main kernel: bf16 K/V from workspace ----------------
__global__ __launch_bounds__(256) void fa_fwd_bf16_kernel(const float* __restrict__ Q,
                                                          const __bf16* __restrict__ Kb,
                                                          const __bf16* __restrict__ Vb,
                                                          float* __restrict__ O) {
  const int lane = threadIdx.x & 31;
  const int wave = threadIdx.x >> 5;
  const int head = blockIdx.y;
  const int q0   = blockIdx.x * kQTile + wave * 16;
  const int half = lane >> 4;
  const int l16  = lane & 15;
  const int koff = half ? 8 : 0;

  // 32KB LDS: double-buffered K/V blocks; reused first for Q-transpose staging.
  __shared__ __bf16 sbuf[2][2][32 * kD];

  // ---- stage this wave's Q (16x128) transposed+scaled into its slice of sbuf ----
  __bf16* qstage = &sbuf[0][0][0] + wave * (16 * kD);
  for (int idx = lane; idx < 16 * kD; idx += 32) {
    const int qr = idx >> 7, d = idx & (kD - 1);
    const float v = Q[((size_t)(q0 + qr) * kH + head) * kD + d];
    qstage[d * 16 + qr] = (__bf16)(v * kScale);
  }
  __syncthreads();

  // ---- Q^T as four 32(d)x16(q) B-fragments: lane l holds column q = l%16, d in table order ----
  bf16x16 qfrag[4];
#pragma unroll
  for (int c = 0; c < 4; ++c) {
    Frag qa;
    const __bf16* qp = qstage + (c * 32 + koff) * 16 + l16;
#pragma unroll
    for (int i = 0; i < 8; ++i) {
      qa.v[i]     = qp[i * 16];
      qa.v[i + 8] = qp[(16 + i) * 16];
    }
    qfrag[c] = qa.v;
  }
  __syncthreads();   // Q reads done; sbuf now owned by the K/V pipeline

  f32x8 oacc[8];
  {
    f32x8 z = {};
#pragma unroll
    for (int n = 0; n < 8; ++n) oacc[n] = z;
  }
  float m = -__builtin_inff(), lsum = 0.f;

  stage_kv(Kb, Vb, &sbuf[0][0][0], &sbuf[0][1][0], 0, head, threadIdx.x);

#pragma unroll 1
  for (int blk = 0; blk < kS / 32; ++blk) {
    const int cur = blk & 1, nxt = cur ^ 1;
    if (blk + 1 < kS / 32) {
      stage_kv(Kb, Vb, &sbuf[nxt][0][0], &sbuf[nxt][1][0], (blk + 1) * 32, head, threadIdx.x);
      wait_async_le(4);            // current block's 4 async ops drained
    } else {
      wait_async_le(0);
    }
    __syncthreads();               // current buffers fully written by all threads

    const __bf16* kcur = &sbuf[cur][0][0];
    const __bf16* vcur = &sbuf[cur][1][0];

    // ---- S^T = K_blk . Q^T : two 16(kv)x16(q) f32 tiles, contract over D=128 ----
    f32x8 sA = {}, sB = {};
#pragma unroll
    for (int c = 0; c < 4; ++c) {
      Frag ka, kb;
      const __bf16* kp0 = kcur + l16 * kD + c * 32 + koff;          // kv tile 0
      const __bf16* kp1 = kp0 + 16 * kD;                            // kv tile 1
      ka.u[0] = *(const uint4*)kp0;  ka.u[1] = *(const uint4*)(kp0 + 16);
      kb.u[0] = *(const uint4*)kp1;  kb.u[1] = *(const uint4*)(kp1 + 16);
      sA = WMMA_BF16(ka.v, qfrag[c], sA);
      sB = WMMA_BF16(kb.v, qfrag[c], sB);
    }

    // ---- online softmax merge (base-2 domain; scale folded into Q) ----
    float bm = sA[0];
#pragma unroll
    for (int i = 1; i < 8; ++i) bm = fmaxf(bm, sA[i]);
#pragma unroll
    for (int i = 0; i < 8; ++i) bm = fmaxf(bm, sB[i]);
    bm = fmaxf(bm, __shfl_xor(bm, 16));
    const float nm    = fmaxf(m, bm);
    const float alpha = exp2f(m - nm);

    bf16x16 pfrag;                 // P A-fragment: pure in-lane repack
    float rs = 0.f;
#pragma unroll
    for (int i = 0; i < 8; ++i) {
      const float pa = exp2f(sA[i] - nm);
      const float pb = exp2f(sB[i] - nm);
      rs += pa + pb;
      pfrag[i]     = (__bf16)pa;
      pfrag[i + 8] = (__bf16)pb;
    }
    rs += __shfl_xor(rs, 16);
    lsum = lsum * alpha + rs;
    m    = nm;
#pragma unroll
    for (int n = 0; n < 8; ++n) oacc[n] *= alpha;

    // ---- O += P . V : 8 d-slices; B-fragment column d = n*16 + l%16, kv in table order ----
#pragma unroll
    for (int n = 0; n < 8; ++n) {
      Frag vf;
      const __bf16* vp = vcur + koff * kD + n * 16 + l16;
#pragma unroll
      for (int i = 0; i < 8; ++i) {
        vf.v[i]     = vp[i * kD];
        vf.v[i + 8] = vp[(16 + i) * kD];
      }
      oacc[n] = WMMA_BF16(pfrag, vf.v, oacc[n]);
    }
    __syncthreads();               // all reads of buf[cur] done before it is re-staged
  }

  // ---- epilogue: normalize and store fp32, coalesced per 16-lane half ----
#pragma unroll
  for (int r = 0; r < 8; ++r) {
    const int qr = r + half * 8;
    const float inv = 1.0f / __shfl(lsum, qr, 32);
    const size_t ob = ((size_t)(q0 + qr) * kH + head) * kD + l16;
#pragma unroll
    for (int n = 0; n < 8; ++n) O[ob + n * 16] = oacc[n][r] * inv;
  }
}

// ---------------- fallback (round-1): fp32 loads + in-loop convert ----------------
__global__ __launch_bounds__(256) void fa_fwd_f32_kernel(const float* __restrict__ Q,
                                                         const float* __restrict__ K,
                                                         const float* __restrict__ V,
                                                         float* __restrict__ O) {
  const int lane = threadIdx.x & 31;
  const int wave = threadIdx.x >> 5;
  const int head = blockIdx.y;
  const int q0   = blockIdx.x * kQTile + wave * 16;
  const int half = lane >> 4;
  const int l16  = lane & 15;
  const int koff = half ? 8 : 0;

  __shared__ __bf16 qt[kWaves][kD][16];
  for (int idx = lane; idx < 16 * kD; idx += 32) {
    const int qr = idx >> 7, d = idx & (kD - 1);
    const float v = Q[((size_t)(q0 + qr) * kH + head) * kD + d];
    qt[wave][d][qr] = (__bf16)(v * kScale);
  }
  __syncthreads();

  bf16x16 qfrag[4];
#pragma unroll
  for (int c = 0; c < 4; ++c) {
#pragma unroll
    for (int i = 0; i < 8; ++i) {
      qfrag[c][i]     = qt[wave][c * 32 + koff + i][l16];
      qfrag[c][i + 8] = qt[wave][c * 32 + 16 + koff + i][l16];
    }
  }

  f32x8 oacc[8];
  {
    f32x8 z = {};
#pragma unroll
    for (int n = 0; n < 8; ++n) oacc[n] = z;
  }
  float m = -__builtin_inff(), lsum = 0.f;

#pragma unroll 1
  for (int s0 = 0; s0 < kS; s0 += 32) {
    const float* k0p = K + ((size_t)(s0 + l16) * kH + head) * kD;
    const float* k1p = k0p + (size_t)16 * kH * kD;
    f32x8 sA = {}, sB = {};
#pragma unroll
    for (int c = 0; c < 4; ++c) {
      bf16x16 ka, kb;
#pragma unroll
      for (int i = 0; i < 8; ++i) {
        ka[i]     = (__bf16)k0p[c * 32 + koff + i];
        ka[i + 8] = (__bf16)k0p[c * 32 + 16 + koff + i];
        kb[i]     = (__bf16)k1p[c * 32 + koff + i];
        kb[i + 8] = (__bf16)k1p[c * 32 + 16 + koff + i];
      }
      sA = WMMA_BF16(ka, qfrag[c], sA);
      sB = WMMA_BF16(kb, qfrag[c], sB);
    }

    float bm = sA[0];
#pragma unroll
    for (int i = 1; i < 8; ++i) bm = fmaxf(bm, sA[i]);
#pragma unroll
    for (int i = 0; i < 8; ++i) bm = fmaxf(bm, sB[i]);
    bm = fmaxf(bm, __shfl_xor(bm, 16));
    const float nm    = fmaxf(m, bm);
    const float alpha = exp2f(m - nm);

    bf16x16 pfrag;
    float rs = 0.f;
#pragma unroll
    for (int i = 0; i < 8; ++i) {
      const float pa = exp2f(sA[i] - nm);
      const float pb = exp2f(sB[i] - nm);
      rs += pa + pb;
      pfrag[i]     = (__bf16)pa;
      pfrag[i + 8] = (__bf16)pb;
    }
    rs += __shfl_xor(rs, 16);
    lsum = lsum * alpha + rs;
    m    = nm;
#pragma unroll
    for (int n = 0; n < 8; ++n) oacc[n] *= alpha;

    const float* vp = V + ((size_t)(s0 + koff) * kH + head) * kD + l16;
#pragma unroll
    for (int n = 0; n < 8; ++n) {
      bf16x16 vf;
#pragma unroll
      for (int i = 0; i < 8; ++i) {
        vf[i]     = (__bf16)vp[(size_t)i * kH * kD + n * 16];
        vf[i + 8] = (__bf16)vp[(size_t)(16 + i) * kH * kD + n * 16];
      }
      oacc[n] = WMMA_BF16(pfrag, vf, oacc[n]);
    }
  }

#pragma unroll
  for (int r = 0; r < 8; ++r) {
    const int qr = r + half * 8;
    const float inv = 1.0f / __shfl(lsum, qr, 32);
    const size_t ob = ((size_t)(q0 + qr) * kH + head) * kD + l16;
#pragma unroll
    for (int n = 0; n < 8; ++n) O[ob + n * 16] = oacc[n][r] * inv;
  }
}

extern "C" void kernel_launch(void* const* d_in, const int* in_sizes, int n_in,
                              void* d_out, int out_size, void* d_ws, size_t ws_size,
                              hipStream_t stream) {
  const float* Q = (const float*)d_in[0];
  const float* K = (const float*)d_in[1];
  const float* V = (const float*)d_in[2];
  float*       O = (float*)d_out;
  (void)in_sizes; (void)n_in; (void)out_size;

  const size_t n    = (size_t)kS * kH * kD;       // 8M elements per tensor
  const size_t need = n * sizeof(__bf16) * 2;     // bf16 K + V = 32 MB
  dim3 grid(kS / kQTile, kH);                     // (32, 16)

  if (ws_size >= need) {
    __bf16* Kb = (__bf16*)d_ws;
    __bf16* Vb = Kb + n;
    const int n8 = (int)(n / 8);
    cvt_bf16_kernel<<<dim3((n8 + 255) / 256), dim3(256), 0, stream>>>(K, Kb, n8);
    cvt_bf16_kernel<<<dim3((n8 + 255) / 256), dim3(256), 0, stream>>>(V, Vb, n8);
    fa_fwd_bf16_kernel<<<grid, dim3(256), 0, stream>>>(Q, Kb, Vb, O);
  } else {
    fa_fwd_f32_kernel<<<grid, dim3(256), 0, stream>>>(Q, K, V, O);
  }
}
